// LongcatMoe_48421461295449
// MI455X (gfx1250) — matrix operand order repaired and checked by reading, not compiled, not run on hardware
//
#include <hip/hip_runtime.h>
#include <hip/hip_bf16.h>

// ---------------- problem constants ----------------
constexpr int T_ = 1024;   // tokens
constexpr int H_ = 2048;   // hidden
constexpr int I_ = 1024;   // moe intermediate
constexpr int E_ = 8;      // routed experts
constexpr int NE_ = 16;    // routed + zero experts
constexpr float ROUTED_SCALING = 1.0f;

// ---------------- WMMA types ----------------
typedef __attribute__((ext_vector_type(16))) __bf16 v16bf;
typedef __attribute__((ext_vector_type(8)))  float  v8f;
typedef __attribute__((ext_vector_type(2)))  float  vf2;
typedef __attribute__((ext_vector_type(2)))  __bf16 vbf2;

union FragBF { unsigned u[8]; v16bf v; };

// single v_cvt_pk_bf16_f32
__device__ __forceinline__ unsigned pack2(float a, float b) {
  vf2 f; f.x = a; f.y = b;
  vbf2 h = __builtin_convertvector(f, vbf2);
  return __builtin_bit_cast(unsigned, h);
}
__device__ __forceinline__ unsigned short bf(float f) {
  __bf16 h = (__bf16)f;
  return __builtin_bit_cast(unsigned short, h);
}
__device__ __forceinline__ v8f vzero8() {
  v8f z = {0.f,0.f,0.f,0.f,0.f,0.f,0.f,0.f};
  return z;
}

// A fragment: 16x32 bf16 tile, row-major in LDS (row stride 32).
// lanes 0-15 -> M=lane, K in {0..7,16..23}; lanes 16-31 -> M=lane-16, K in {8..15,24..31}
__device__ __forceinline__ v16bf load_frag_a(const unsigned short* lds, int lane) {
  FragBF f;
  const int row  = lane & 15;
  const int koff = (lane >> 4) << 3;     // 0 or 8
  const unsigned short* base = lds + row * 32;
#pragma unroll
  for (int v = 0; v < 4; ++v)
    f.u[v] = *(const unsigned*)(base + koff + 2 * v);
#pragma unroll
  for (int v = 0; v < 4; ++v)
    f.u[4 + v] = *(const unsigned*)(base + 16 + koff + 2 * v);
  return f.v;
}

// B fragment: 32x16 bf16 tile stored transposed in LDS as [n][k] (row stride 32).
// lanes 0-15 -> N=lane, K in {0..7,16..23} ; lanes 16-31 -> N=lane-16, K in {8..15,24..31}
__device__ __forceinline__ v16bf load_frag_b(const unsigned short* lds, int lane) {
  FragBF f;
  const int n     = lane & 15;
  const int kbase = (lane >> 4) << 3;    // 0 or 8
  const unsigned short* base = lds + n * 32 + kbase;
#pragma unroll
  for (int v = 0; v < 4; ++v)
    f.u[v] = *(const unsigned*)(base + 2 * v);
#pragma unroll
  for (int v = 0; v < 4; ++v)
    f.u[4 + v] = *(const unsigned*)(base + 16 + 2 * v);
  return f.v;
}

// ---------------- router: softmax + top-4 selection ----------------
__global__ __launch_bounds__(256)
void moe_router(const float* __restrict__ x, const float* __restrict__ rw_mat,
                const float* __restrict__ bias, float* __restrict__ route_w,
                float* __restrict__ zero_w) {
  __shared__ float red[256];
  __shared__ float logits[NE_];
  const int t   = blockIdx.x;
  const int tid = threadIdx.x;
  const int g   = tid >> 4;              // expert 0..15
  const int j   = tid & 15;              // reducer lane
  const float* xr = x + (size_t)t * H_;
  const float* wr = rw_mat + (size_t)g * H_;
  float p = 0.f;
  for (int h = j; h < H_; h += 16) p += xr[h] * wr[h];
  red[tid] = p;
  __syncthreads();
  if (tid < NE_) {
    float s = 0.f;
#pragma unroll
    for (int k = 0; k < 16; ++k) s += red[tid * 16 + k];
    logits[tid] = s;
  }
  __syncthreads();
  if (tid == 0) {
    float l[NE_], sc[NE_], sel[NE_];
    float mx = -1e30f;
#pragma unroll
    for (int i = 0; i < NE_; ++i) { l[i] = logits[i]; mx = fmaxf(mx, l[i]); }
    float sum = 0.f;
#pragma unroll
    for (int i = 0; i < NE_; ++i) { sc[i] = __expf(l[i] - mx); sum += sc[i]; }
    const float inv = 1.f / sum;
#pragma unroll
    for (int i = 0; i < NE_; ++i) { sc[i] *= inv; sel[i] = sc[i] + bias[i]; }
    float rw[E_];
#pragma unroll
    for (int i = 0; i < E_; ++i) rw[i] = 0.f;
    float zw = 0.f;
#pragma unroll
    for (int k = 0; k < 4; ++k) {        // TOP_K = 4, argmax w/ masking
      int best = 0; float bv = -1e30f;
#pragma unroll
      for (int i = 0; i < NE_; ++i)
        if (sel[i] > bv) { bv = sel[i]; best = i; }
      sel[best] = -1e30f;
      const float w = sc[best] * ROUTED_SCALING;
      if (best < E_) rw[best] += w; else zw += w;
    }
#pragma unroll
    for (int i = 0; i < E_; ++i) route_w[t * E_ + i] = rw[i];
    zero_w[t] = zw;
  }
}

// ---------------- init: out = zero_w[t] * x ----------------
__global__ __launch_bounds__(256)
void moe_init_out(const float* __restrict__ x, const float* __restrict__ zero_w,
                  float* __restrict__ out) {
  const size_t idx  = ((size_t)blockIdx.x * 256 + threadIdx.x) * 4;
  const int    t    = (int)(idx / H_);
  const float  zw   = zero_w[t];
  const float4 v    = *(const float4*)(x + idx);
  float4 o;
  o.x = zw * v.x; o.y = zw * v.y; o.z = zw * v.z; o.w = zw * v.w;
  *(float4*)(out + idx) = o;
}

// ---------------- gate/up GEMM + SiLU -> mid (bf16) ----------------
// grid: (I/64, T/128, E), block 256 (8 waves); 2-stage software pipeline.
// B staging: each thread owns 2x(2k x 2n) blocks -> coalesced float2 loads,
// packed K-pair converts (v_cvt_pk_bf16_f32), ds_store_b32 into [n][k] tile.
__global__ __launch_bounds__(256)
void moe_gateup(const float* __restrict__ x, const float* __restrict__ w_gate,
                const float* __restrict__ w_up, unsigned short* __restrict__ mid) {
  __shared__ unsigned short As[2][128 * 32];
  __shared__ unsigned short Bg[2][64 * 32];
  __shared__ unsigned short Bu[2][64 * 32];

  const int e    = blockIdx.z;
  const int n0   = blockIdx.x * 64;
  const int t0   = blockIdx.y * 128;
  const int tid  = threadIdx.x;
  const int lane = tid & 31;
  const int wave = tid >> 5;

  const float* wg = w_gate + (size_t)e * H_ * I_;
  const float* wu = w_up   + (size_t)e * H_ * I_;

  v8f accG[4], accU[4];
#pragma unroll
  for (int i = 0; i < 4; ++i) { accG[i] = vzero8(); accU[i] = vzero8(); }

  float4 ar[4];                          // A: next K tile (fp32)
  float2 g0r[2], g1r[2], u0r[2], u1r[2]; // B: rows k, k+1 of a 2x2 block

  auto load_regs = [&](int k0) {
#pragma unroll
    for (int i = 0; i < 4; ++i) {        // A: 128x32 fp32 = 1024 float4's
      const int idx = tid + 256 * i;
      const int row = idx >> 3;
      const int seg = idx & 7;
      ar[i] = *(const float4*)(x + (size_t)(t0 + row) * H_ + k0 + seg * 4);
    }
#pragma unroll
    for (int i = 0; i < 2; ++i) {        // B: 16 kpairs x 32 npairs = 512 blocks
      const int idx = tid + 256 * i;
      const int kp  = idx >> 5;          // 0..15
      const int np  = idx & 31;          // 0..31
      const size_t off = (size_t)(k0 + 2 * kp) * I_ + n0 + 2 * np;
      g0r[i] = *(const float2*)(wg + off);
      g1r[i] = *(const float2*)(wg + off + I_);
      u0r[i] = *(const float2*)(wu + off);
      u1r[i] = *(const float2*)(wu + off + I_);
    }
  };
  auto store_lds = [&](int buf) {
#pragma unroll
    for (int i = 0; i < 4; ++i) {
      const int idx = tid + 256 * i;
      const int row = idx >> 3;
      const int seg = idx & 7;
      unsigned* dst = (unsigned*)(&As[buf][row * 32 + seg * 4]);
      dst[0] = pack2(ar[i].x, ar[i].y);
      dst[1] = pack2(ar[i].z, ar[i].w);
    }
#pragma unroll
    for (int i = 0; i < 2; ++i) {        // pack along K, store [n][k] pairs
      const int idx = tid + 256 * i;
      const int kp  = idx >> 5;
      const int np  = idx & 31;
      const int n   = 2 * np;
      const int kk  = 2 * kp;
      *(unsigned*)(&Bg[buf][(n + 0) * 32 + kk]) = pack2(g0r[i].x, g1r[i].x);
      *(unsigned*)(&Bg[buf][(n + 1) * 32 + kk]) = pack2(g0r[i].y, g1r[i].y);
      *(unsigned*)(&Bu[buf][(n + 0) * 32 + kk]) = pack2(u0r[i].x, u1r[i].x);
      *(unsigned*)(&Bu[buf][(n + 1) * 32 + kk]) = pack2(u0r[i].y, u1r[i].y);
    }
  };

  load_regs(0);
  store_lds(0);
  __syncthreads();

  for (int k0 = 0; k0 < H_; k0 += 32) {
    const int  cur  = (k0 >> 5) & 1;
    const bool more = (k0 + 32) < H_;
    if (more) {
      load_regs(k0 + 32);                // in flight during WMMAs below
      if (k0 + 64 < H_) {
        const size_t poff = (size_t)(k0 + 64 + (tid >> 3)) * I_ + n0 + (tid & 7) * 8;
        __builtin_prefetch(wg + poff, 0, 3);
        __builtin_prefetch(wu + poff, 0, 3);
      }
    }
    const v16bf a = load_frag_a(&As[cur][wave * 16 * 32], lane);
#pragma unroll
    for (int nt = 0; nt < 4; ++nt) {
      const v16bf bgf = load_frag_b(&Bg[cur][nt * 16 * 32], lane);
      const v16bf buf = load_frag_b(&Bu[cur][nt * 16 * 32], lane);
      accG[nt] = __builtin_amdgcn_wmma_f32_16x16x32_bf16(
          false, a, false, bgf, (short)0, accG[nt], false, false);
      accU[nt] = __builtin_amdgcn_wmma_f32_16x16x32_bf16(
          false, a, false, buf, (short)0, accU[nt], false, false);
    }
    if (more) store_lds(cur ^ 1);        // loadcnt wait lands here, post-WMMA
    __syncthreads();
  }

  // epilogue: mid = silu(gate) * up -> bf16 global
  const int n_lane = lane & 15;
  const int m_hi   = (lane >> 4) * 8;
  unsigned short* me = mid + (size_t)e * T_ * I_;
#pragma unroll
  for (int nt = 0; nt < 4; ++nt) {
#pragma unroll
    for (int v = 0; v < 8; ++v) {
      const float g = accG[nt][v];
      const float u = accU[nt][v];
      const float m = (g / (1.0f + __expf(-g))) * u;   // SiLU(g)*u
      const int t = t0 + wave * 16 + m_hi + v;
      const int i = n0 + nt * 16 + n_lane;
      me[(size_t)t * I_ + i] = bf(m);
    }
  }
}

// ---------------- down GEMM + weighted combine ----------------
// grid: (H/64, T/128, E); same 2-stage pipeline; accumulate with f32 atomics
__global__ __launch_bounds__(256)
void moe_down(const unsigned short* __restrict__ mid, const float* __restrict__ w_down,
              const float* __restrict__ route_w, float* __restrict__ out) {
  __shared__ unsigned short As[2][128 * 32];
  __shared__ unsigned short Bd[2][64 * 32];

  const int e    = blockIdx.z;
  const int n0   = blockIdx.x * 64;      // over H
  const int t0   = blockIdx.y * 128;
  const int tid  = threadIdx.x;
  const int lane = tid & 31;
  const int wave = tid >> 5;

  const unsigned short* me = mid + (size_t)e * T_ * I_;
  const float* wd = w_down + (size_t)e * I_ * H_;

  v8f acc[4];
#pragma unroll
  for (int i = 0; i < 4; ++i) acc[i] = vzero8();

  uint4  ar[2];                          // A already bf16: raw copy
  float2 d0r[2], d1r[2];                 // B 2x2 blocks

  auto load_regs = [&](int k0) {
#pragma unroll
    for (int i = 0; i < 2; ++i) {        // A: 128x32 bf16 = 512 uint4's
      const int idx = tid + 256 * i;
      const int row = idx >> 2;
      const int seg = idx & 3;           // uint4 = 8 bf16
      ar[i] = *(const uint4*)(me + (size_t)(t0 + row) * I_ + k0 + seg * 8);
    }
#pragma unroll
    for (int i = 0; i < 2; ++i) {        // B: 16 kpairs x 32 npairs
      const int idx = tid + 256 * i;
      const int kp  = idx >> 5;
      const int np  = idx & 31;
      const size_t off = (size_t)(k0 + 2 * kp) * H_ + n0 + 2 * np;
      d0r[i] = *(const float2*)(wd + off);
      d1r[i] = *(const float2*)(wd + off + H_);
    }
  };
  auto store_lds = [&](int buf) {
#pragma unroll
    for (int i = 0; i < 2; ++i) {
      const int idx = tid + 256 * i;
      const int row = idx >> 2;
      const int seg = idx & 3;
      *(uint4*)(&As[buf][row * 32 + seg * 8]) = ar[i];
    }
#pragma unroll
    for (int i = 0; i < 2; ++i) {        // pack along K, store [n][k] pairs
      const int idx = tid + 256 * i;
      const int kp  = idx >> 5;
      const int np  = idx & 31;
      const int n   = 2 * np;
      const int kk  = 2 * kp;
      *(unsigned*)(&Bd[buf][(n + 0) * 32 + kk]) = pack2(d0r[i].x, d1r[i].x);
      *(unsigned*)(&Bd[buf][(n + 1) * 32 + kk]) = pack2(d0r[i].y, d1r[i].y);
    }
  };

  load_regs(0);
  store_lds(0);
  __syncthreads();

  for (int k0 = 0; k0 < I_; k0 += 32) {
    const int  cur  = (k0 >> 5) & 1;
    const bool more = (k0 + 32) < I_;
    if (more) {
      load_regs(k0 + 32);
      if (k0 + 64 < I_)
        __builtin_prefetch(wd + (size_t)(k0 + 64 + (tid >> 3)) * H_ + n0 + (tid & 7) * 8, 0, 3);
    }
    const v16bf a = load_frag_a(&As[cur][wave * 16 * 32], lane);
#pragma unroll
    for (int nt = 0; nt < 4; ++nt) {
      const v16bf b = load_frag_b(&Bd[cur][nt * 16 * 32], lane);
      acc[nt] = __builtin_amdgcn_wmma_f32_16x16x32_bf16(
          false, a, false, b, (short)0, acc[nt], false, false);
    }
    if (more) store_lds(cur ^ 1);
    __syncthreads();
  }

  // epilogue: out[t,h] += route_w[t,e] * acc  (skip zero routes)
  const int n_lane = lane & 15;
  const int m_hi   = (lane >> 4) * 8;
  const int tbase  = t0 + wave * 16 + m_hi;
  float rw[8];
#pragma unroll
  for (int v = 0; v < 8; ++v) rw[v] = route_w[(size_t)(tbase + v) * E_ + e];
#pragma unroll
  for (int nt = 0; nt < 4; ++nt) {
    const int h = n0 + nt * 16 + n_lane;
#pragma unroll
    for (int v = 0; v < 8; ++v) {
      if (rw[v] != 0.f)
        atomicAdd(out + (size_t)(tbase + v) * H_ + h, rw[v] * acc[nt][v]);
    }
  }
}

// ---------------- launch ----------------
extern "C" void kernel_launch(void* const* d_in, const int* in_sizes, int n_in,
                              void* d_out, int out_size, void* d_ws, size_t ws_size,
                              hipStream_t stream) {
  const float* x         = (const float*)d_in[0];   // [T,H]
  const float* router_w  = (const float*)d_in[1];   // [E+Z,H]
  const float* corr_bias = (const float*)d_in[2];   // [E+Z]
  const float* w_gate    = (const float*)d_in[3];   // [E,H,I]
  const float* w_up      = (const float*)d_in[4];   // [E,H,I]
  const float* w_down    = (const float*)d_in[5];   // [E,I,H]
  float* out = (float*)d_out;                       // [T,H] fp32

  // workspace layout
  float* route_w = (float*)d_ws;                    // T*E fp32   (32 KB)
  float* zero_w  = route_w + (size_t)T_ * E_;       // T fp32     (4 KB)
  unsigned short* mid = (unsigned short*)(zero_w + T_); // [E,T,I] bf16 (16 MB)

  moe_router<<<T_, 256, 0, stream>>>(x, router_w, corr_bias, route_w, zero_w);
  moe_init_out<<<(T_ * H_ / 4) / 256, 256, 0, stream>>>(x, zero_w, out);

  dim3 g1(I_ / 64, T_ / 128, E_);
  moe_gateup<<<g1, 256, 0, stream>>>(x, w_gate, w_up, mid);

  dim3 g2(H_ / 64, T_ / 128, E_);
  moe_down<<<g2, 256, 0, stream>>>(mid, w_down, route_w, out);
}
